// Quantizer_69965017251885
// MI455X (gfx1250) — matrix operand order repaired and checked by reading, not compiled, not run on hardware
//
#include <hip/hip_runtime.h>
#include <hip/hip_bf16.h>
#include <float.h>

typedef __attribute__((ext_vector_type(16))) _Float16 v16h;
typedef __attribute__((ext_vector_type(8)))  float    v8f;
typedef __attribute__((ext_vector_type(4)))  unsigned int u32x4;
typedef __attribute__((ext_vector_type(8)))  int      i32x8;
typedef __attribute__((ext_vector_type(4)))  int      i32x4;

#define N_VEC   65536
#define KSIZE   1024
#define BETA    0.25f
#define DECAY   0.99f
#define EPSV    1e-5f

// workspace byte offsets
#define XROT_OFF   0u          // 65536*64 f32 = 16777216 B
#define XR16_OFF   16777216u   // 65536*64 f16 = 8388608 B
#define EF16_OFF   25165824u   // 1024*64 f16  = 131072 B
#define ENORM_OFF  25296896u   // 1024 f32     = 4096 B
#define IDXWS_OFF  25300992u   // 65536 i32    = 262144 B
#define COUNTS_OFF 25563136u   // 1024 f32
#define DW_OFF     25567232u   // 65536 f32
#define MSE_OFF    25829376u   // 1 f32
#define ACC_BYTES  (4096u + 262144u + 4u)

// d_out float-element offsets (outputs concatenated in return order)
#define OUT_QUANT 0
#define OUT_LOSS  4194304
#define OUT_IDX   4194305
#define OUT_NEWE  4259841
#define OUT_NCS   4325377
#define OUT_NEMW  4326401

// order-preserving f32 -> u32 (no NaNs in this pipeline)
__device__ __forceinline__ unsigned int f32_ordered(float v) {
  unsigned int b = __float_as_uint(v);
  return ((int)b < 0) ? ~b : (b | 0x80000000u);
}

// ---------------- x_rot = x_flat @ R  (f32 exact + f16 copy for TDM) -------
__global__ void __launch_bounds__(256)
rotate_kernel(const float* __restrict__ x, const float* __restrict__ R,
              float* __restrict__ xrot, _Float16* __restrict__ xr16) {
  __shared__ float xs[64 * 64];   // [c][row]
  __shared__ float rs[64 * 64];   // [c][j]
  const int t  = threadIdx.x;
  const int n0 = blockIdx.x * 64;         // 64 rows per block; stays in one b
  const int b  = n0 >> 12;
  const int hw0 = n0 & 4095;
  const float* xb = x + (size_t)b * 262144 + hw0;
#pragma unroll
  for (int i = 0; i < 16; ++i) {
    int e = t + i * 256;
    int c = e >> 6, r = e & 63;
    xs[e] = xb[(size_t)c * 4096 + r];     // coalesced per c
    rs[e] = R[e];
  }
  __syncthreads();
  const int j  = t & 63;                  // output column
  const int rg = t >> 6;                  // row group (16 rows)
  float acc[16];
#pragma unroll
  for (int rr = 0; rr < 16; ++rr) acc[rr] = 0.f;
  for (int c = 0; c < 64; ++c) {
    float rv = rs[c * 64 + j];
#pragma unroll
    for (int rr = 0; rr < 16; ++rr)
      acc[rr] += xs[c * 64 + rg * 16 + rr] * rv;
  }
#pragma unroll
  for (int rr = 0; rr < 16; ++rr) {
    size_t o = (size_t)(n0 + rg * 16 + rr) * 64 + j;
    xrot[o] = acc[rr];
    xr16[o] = (_Float16)acc[rr];
  }
}

// ---------------- codebook: f16 copy + squared norms -----------------------
__global__ void prep_codebook(const float* __restrict__ E,
                              _Float16* __restrict__ ef16,
                              float* __restrict__ enorm) {
  int k = blockIdx.x * blockDim.x + threadIdx.x;
  if (k >= KSIZE) return;
  float s = 0.f;
#pragma unroll 8
  for (int c = 0; c < 64; ++c) {
    float e = E[k * 64 + c];
    s += e * e;
    ef16[k * 64 + c] = (_Float16)e;
  }
  enorm[k] = s;
}

// ---------------- WMMA distance GEMM + fused argmin ------------------------
// block = 8 waves; 64 rows x full K=1024.
// Each wave: 8 k-tiles of 16 cols x 4 row-tiles of 16 rows (B regs reused 4x).
// A tile (64x64 f16) staged global->LDS by the Tensor Data Mover.
// Argmin kept branchless via packed u64 (ordered_dist<<32 | k) min-reduction.
__global__ void __launch_bounds__(256)
dist_argmin(const _Float16* __restrict__ xr16, const _Float16* __restrict__ ef16,
            const float* __restrict__ enorm, int* __restrict__ idx_ws,
            float* __restrict__ out_idx) {
  __shared__ _Float16 xs[64 * 64];                 // A rows f16, [row][d], 8KB
  __shared__ unsigned long long redk[8 * 64];      // per-wave row keys
  const int t  = threadIdx.x;
  const int n0 = blockIdx.x * 64;

  // ---- TDM: stage 64x64 f16 tile (rows n0..n0+63) into LDS ----
  if (t < 32) {
    unsigned long long gaddr =
        (unsigned long long)(uintptr_t)(xr16) + (unsigned long long)n0 * 128ull;
    unsigned int ldsa = (unsigned int)(uintptr_t)(&xs[0]);
    u32x4 g0;
    g0.x = 0x1u;                                   // count=1 valid descriptor
    g0.y = ldsa;                                   // lds_addr
    g0.z = (unsigned int)(gaddr & 0xFFFFFFFFull);  // global_addr[31:0]
    g0.w = ((unsigned int)(gaddr >> 32) & 0x01FFFFFFu) | (2u << 30); // type=2
    i32x8 g1;
    g1[0] = 0x00010000;      // workgroup_mask=0, data_size=1 (2 bytes)
    g1[1] = 64 << 16;        // tensor_dim0 = 64 (low 16 bits)
    g1[2] = 64 << 16;        // tensor_dim0 hi=0 | tensor_dim1 lo = 64
    g1[3] = 64 << 16;        // tensor_dim1 hi=0 | tile_dim0 = 64
    g1[4] = 64;              // tile_dim1 = 64, tile_dim2 = 0
    g1[5] = 64;              // tensor_dim0_stride = 64
    g1[6] = 0;               // stride0 hi | stride1 lo
    g1[7] = 0;               // stride1 hi
    i32x4 gz = {0, 0, 0, 0};
#if __clang_major__ >= 23
    i32x8 gz8 = {0, 0, 0, 0, 0, 0, 0, 0};
    __builtin_amdgcn_tensor_load_to_lds(g0, g1, gz, gz, gz8, 0);
#else
    __builtin_amdgcn_tensor_load_to_lds(g0, g1, gz, gz, 0);
#endif
    __builtin_amdgcn_s_wait_tensorcnt(0);
  }
  __syncthreads();
  asm volatile("" ::: "memory");

  const int wave = t >> 5;
  const int lane = t & 31;
  const int m    = lane & 15;
  const int half = lane >> 4;

  // A fragments for 4 row-tiles (persist across all k-tiles).
  // 16x32 f16 layout: lanes 0-15 hold K 0-7/16-23, lanes 16-31 hold K 8-15/24-31.
  v16h a0[4], a1[4];
#pragma unroll
  for (int rt = 0; rt < 4; ++rt) {
    int mg = rt * 16 + m;
#pragma unroll
    for (int v = 0; v < 8; ++v) {
      int kk = ((v >> 2) << 4) + (half << 3) + ((v & 3) << 1);
      a0[rt][2 * v]     = xs[mg * 64 + kk];
      a0[rt][2 * v + 1] = xs[mg * 64 + kk + 1];
      a1[rt][2 * v]     = xs[mg * 64 + 32 + kk];
      a1[rt][2 * v + 1] = xs[mg * 64 + 32 + kk + 1];
    }
  }

  unsigned long long best[4][8];
#pragma unroll
  for (int rt = 0; rt < 4; ++rt)
#pragma unroll
    for (int r = 0; r < 8; ++r) best[rt][r] = 0xFFFFFFFFFFFFFFFFull;

  for (int tt = 0; tt < 8; ++tt) {
    const int kbase = (wave * 8 + tt) * 16;
    const int kcol  = kbase + m;           // this lane's output column
    const _Float16* er = ef16 + (size_t)kcol * 64;
    // B fragments (32x16 f16: lanes 0-15 K 0-7/16-23, lanes 16-31 K 8-15/24-31; N = lane&15)
    v16h b0, b1;
#pragma unroll
    for (int v = 0; v < 8; ++v) {
      int d0 = ((v >> 2) << 4) + (half << 3) + ((v & 3) << 1);
      b0[2 * v]     = er[d0];
      b0[2 * v + 1] = er[d0 + 1];
      b1[2 * v]     = er[d0 + 32];
      b1[2 * v + 1] = er[d0 + 33];
    }
    float en = enorm[kcol];
#pragma unroll
    for (int rt = 0; rt < 4; ++rt) {
      v8f c = {};
      c = __builtin_amdgcn_wmma_f32_16x16x32_f16(false, a0[rt], false, b0, (short)0, c, false, false);
      c = __builtin_amdgcn_wmma_f32_16x16x32_f16(false, a1[rt], false, b1, (short)0, c, false, false);
#pragma unroll
      for (int r = 0; r < 8; ++r) {
        float val = fmaf(-2.f, c[r], en);  // row = rt*16 + half*8 + r, col = kcol
        unsigned long long key =
            ((unsigned long long)f32_ordered(val) << 32) | (unsigned int)kcol;
        best[rt][r] = (key < best[rt][r]) ? key : best[rt][r];  // v_cndmask pair
      }
    }
  }
  // reduce argmin across the 16 columns held by each 16-lane group
#pragma unroll
  for (int off = 1; off < 16; off <<= 1) {
#pragma unroll
    for (int rt = 0; rt < 4; ++rt)
#pragma unroll
      for (int r = 0; r < 8; ++r) {
        unsigned long long o = __shfl_xor(best[rt][r], off, 32);
        best[rt][r] = (o < best[rt][r]) ? o : best[rt][r];
      }
  }
  if (m == 0) {
#pragma unroll
    for (int rt = 0; rt < 4; ++rt)
#pragma unroll
      for (int r = 0; r < 8; ++r) {
        int row = rt * 16 + half * 8 + r;
        redk[wave * 64 + row] = best[rt][r];
      }
  }
  __syncthreads();
  if (t < 64) {  // combine the 8 waves' disjoint k-ranges
    unsigned long long bk = redk[t];
#pragma unroll
    for (int w = 1; w < 8; ++w) {
      unsigned long long o = redk[w * 64 + t];
      bk = (o < bk) ? o : bk;
    }
    int bi = (int)(bk & 0xFFFFFFFFull);
    idx_ws[n0 + t]  = bi;
    out_idx[n0 + t] = (float)bi;
  }
}

// ---------------- gather quant, MSE loss, counts, dw -----------------------
__global__ void __launch_bounds__(256)
gather_losses(const float* __restrict__ x, const float* __restrict__ E,
              const float* __restrict__ xrot, const int* __restrict__ idx_ws,
              float* __restrict__ out_quant, float* __restrict__ counts,
              float* __restrict__ dw, float* __restrict__ mse) {
  __shared__ float red[256];
  const int n  = blockIdx.x * blockDim.x + threadIdx.x;
  const int b  = n >> 12, hw = n & 4095;
  const int id = idx_ws[n];
  atomicAdd(&counts[id], 1.f);
  const float* er = E + (size_t)id * 64;
  const float* xb = x + (size_t)b * 262144 + hw;
  float* qb = out_quant + (size_t)b * 262144 + hw;
  float sq = 0.f;
#pragma unroll 4
  for (int c = 0; c < 64; ++c) {
    float e  = er[c];
    float xv = xb[(size_t)c * 4096];
    qb[(size_t)c * 4096] = e;               // coalesced in hw across threads
    float d = e - xv;
    sq += d * d;
    atomicAdd(&dw[id * 64 + c], xrot[(size_t)n * 64 + c]);
  }
  red[threadIdx.x] = sq;
  __syncthreads();
  for (int s = 128; s > 0; s >>= 1) {
    if (threadIdx.x < s) red[threadIdx.x] += red[threadIdx.x + s];
    __syncthreads();
  }
  if (threadIdx.x == 0) atomicAdd(mse, red[0]);
}

// ---------------- entropy, EMA, normalized embedding, loss -----------------
__global__ void __launch_bounds__(1024)
finalize(const float* __restrict__ counts, const float* __restrict__ dw,
         const float* __restrict__ mse, const float* __restrict__ ema_cs,
         const float* __restrict__ ema_w, float* __restrict__ out) {
  __shared__ float red[1024];
  const int k = threadIdx.x;
  float cnt = counts[k];
  float avg = cnt / (float)N_VEC;
  float ent = -avg * logf(avg + 1e-10f);
  float ncs = ema_cs[k] * DECAY + (1.f - DECAY) * cnt;
  red[k] = ncs;
  __syncthreads();
  for (int s = 512; s > 0; s >>= 1) {
    if (k < s) red[k] += red[k + s];
    __syncthreads();
  }
  float ntot = red[0];
  __syncthreads();
  red[k] = ent;
  __syncthreads();
  for (int s = 512; s > 0; s >>= 1) {
    if (k < s) red[k] += red[k + s];
    __syncthreads();
  }
  float entropy = red[0];
  float csf = (ncs + EPSV) / (ntot + (float)KSIZE * EPSV) * ntot;
  out[OUT_NCS + k] = csf;
  for (int c = 0; c < 64; ++c) {
    float nw = ema_w[k * 64 + c] * DECAY + (1.f - DECAY) * dw[k * 64 + c];
    out[OUT_NEMW + k * 64 + c] = nw;
    out[OUT_NEWE + k * 64 + c] = nw / csf;
  }
  if (k == 0)
    out[OUT_LOSS] = (1.f + BETA) * (mse[0] / (float)(N_VEC * 64)) + entropy;
}

extern "C" void kernel_launch(void* const* d_in, const int* in_sizes, int n_in,
                              void* d_out, int out_size, void* d_ws, size_t ws_size,
                              hipStream_t stream) {
  const float* x   = (const float*)d_in[0];
  const float* E   = (const float*)d_in[1];
  const float* R   = (const float*)d_in[2];
  const float* ecs = (const float*)d_in[3];
  const float* emw = (const float*)d_in[4];
  float* out = (float*)d_out;
  char*  ws  = (char*)d_ws;
  float*    xrot   = (float*)(ws + XROT_OFF);
  _Float16* xr16   = (_Float16*)(ws + XR16_OFF);
  _Float16* ef16   = (_Float16*)(ws + EF16_OFF);
  float*    enorm  = (float*)(ws + ENORM_OFF);
  int*      idxws  = (int*)(ws + IDXWS_OFF);
  float*    counts = (float*)(ws + COUNTS_OFF);
  float*    dw     = (float*)(ws + DW_OFF);
  float*    mse    = (float*)(ws + MSE_OFF);

  hipMemsetAsync(ws + COUNTS_OFF, 0, ACC_BYTES, stream);  // zero accumulators every call
  rotate_kernel<<<N_VEC / 64, 256, 0, stream>>>(x, R, xrot, xr16);
  prep_codebook<<<KSIZE / 256, 256, 0, stream>>>(E, ef16, enorm);
  dist_argmin<<<N_VEC / 64, 256, 0, stream>>>(xr16, ef16, enorm, idxws, out + OUT_IDX);
  gather_losses<<<N_VEC / 256, 256, 0, stream>>>(x, E, xrot, idxws,
                                                 out + OUT_QUANT, counts, dw, mse);
  finalize<<<1, 1024, 0, stream>>>(counts, dw, mse, ecs, emw, out);
}
